// LocalAttention_1279900254222
// MI455X (gfx1250) — compile-verified, matrix operand order checked by reading, not run on hardware
//
#include <hip/hip_runtime.h>

// ---------------------------------------------------------------------------
// Problem constants (from reference): B=4, L=2048, D=512, H=8, DK=64, WIN=64
// ---------------------------------------------------------------------------
#define BATCH 4
#define SEQ   2048
#define DMODEL 512
#define NHEAD 8
#define HDIM  64

typedef __attribute__((ext_vector_type(16))) __bf16 v16bf;
typedef __attribute__((ext_vector_type(8)))  float  v8f;
typedef __attribute__((ext_vector_type(4)))  unsigned v4u;
typedef __attribute__((ext_vector_type(8)))  unsigned v8u;

static __device__ __forceinline__ unsigned short bf_bits(__bf16 x) {
    return __builtin_bit_cast(unsigned short, x);
}
static __device__ __forceinline__ __bf16 bits_bf(unsigned short u) {
    return __builtin_bit_cast(__bf16, u);
}

// ---------------------------------------------------------------------------
// WMMA fragment loaders, per CDNA5 ISA 7.12.2 (16-bit A 16x32 / B 32x16).
// A: lane = M%16, half=lane/16; VGPR v holds K pair:
//    K = (v>=4?16:0) + half*8 + (v&3)*2 + {0,1}
// B mirrors A with lane = N when source tile is stored [N][K] row-major.
// ---------------------------------------------------------------------------
static __device__ __forceinline__ v16bf ld_frag_row(const __bf16* p, int ld) {
    const int lane = threadIdx.x & 31;
    const int r  = lane & 15;
    const int lh = lane >> 4;
    const __bf16* row = p + r * ld;
    v16bf f;
#pragma unroll
    for (int v = 0; v < 8; ++v) {
        const int k = ((v & 4) << 2) + (lh << 3) + ((v & 3) << 1);
        f[2 * v]     = row[k];
        f[2 * v + 1] = row[k + 1];
    }
    return f;
}

// Source tile stored [K][N] row-major with row stride ld; lane = column n.
static __device__ __forceinline__ v16bf ld_frag_col(const __bf16* p, int ld) {
    const int lane = threadIdx.x & 31;
    const int n  = lane & 15;
    const int lh = lane >> 4;
    v16bf f;
#pragma unroll
    for (int v = 0; v < 8; ++v) {
        const int k = ((v & 4) << 2) + (lh << 3) + ((v & 3) << 1);
        f[2 * v]     = p[k * ld + n];
        f[2 * v + 1] = p[(k + 1) * ld + n];
    }
    return f;
}

static __device__ __forceinline__ v8f wmma_bf16(v16bf a, v16bf b, v8f c) {
    return __builtin_amdgcn_wmma_f32_16x16x32_bf16(
        /*neg_a=*/false, a, /*neg_b=*/false, b,
        /*c_mod=*/(short)0, c, /*reuse_a=*/false, /*reuse_b=*/false);
}

// ---------------------------------------------------------------------------
// TDM: issue a 2D tensor_load_to_lds.
//   tile: nrows x 64 bf16 elements, tensor row stride 64 elements,
//   LDS rows padded to 72 elements (144 B) via D# pad feature:
//   pad_interval code 4 (=32 DWORDs = 128 B stored), pad_amount code 3 (=4 DWORDs).
// D# layout per CDNA5 ISA 8.3/8.4.  Wave-issued, tracked by TENSORcnt.
// ---------------------------------------------------------------------------
static __device__ __forceinline__ void tdm_load_2d(
    unsigned long long gaddr_bytes, unsigned lds_addr_bytes, int nrows)
{
    v4u g0;
    g0.x = 1u;                                        // count=1 (valid D#)
    g0.y = lds_addr_bytes;                            // lds_addr
    g0.z = (unsigned)(gaddr_bytes & 0xffffffffull);   // global_addr[31:0]
    g0.w = (unsigned)((gaddr_bytes >> 32) & 0x1ffffffull) | (2u << 30); // type=2

    const unsigned nr = (unsigned)nrows & 0xffffu;
    v8u g1;
    g1[0] = (1u << 16)        // data_size = 1 (2 bytes)
          | (1u << 20)        // pad_enable
          | (4u << 22)        // pad_interval: 32 DWORDs
          | (3u << 25);       // pad_amount:   4 DWORDs
    g1[1] = (64u << 16);      // tensor_dim0[15:0] = 64 (bits 63:48)
    g1[2] = (nr << 16);       // tensor_dim0 hi=0 | tensor_dim1[15:0] = nrows
    g1[3] = (64u << 16);      // tensor_dim1 hi=0 | tile_dim0 = 64
    g1[4] = nr;               // tile_dim1 = nrows | tile_dim2 = 0
    g1[5] = 64u;              // tensor_dim0_stride[31:0] = 64 elements
    g1[6] = 0u;               // stride0 hi | stride1 lo
    g1[7] = 0u;               // stride1 hi

    asm volatile("tensor_load_to_lds %0, %1" :: "s"(g0), "s"(g1) : "memory");
}

// ---------------------------------------------------------------------------
// Kernel 0: zero the attn-mean plane (B*L*L floats) with float4 stores.
// ---------------------------------------------------------------------------
__global__ void la_zero_kernel(float4* __restrict__ p, int n4) {
    int i = blockIdx.x * blockDim.x + threadIdx.x;
    const int stride = gridDim.x * blockDim.x;
    for (; i < n4; i += stride) p[i] = make_float4(0.f, 0.f, 0.f, 0.f);
}

// ---------------------------------------------------------------------------
// Kernel 1: fused QKV projection GEMM.  Y = X @ W^T + b, X:[8192,512] f32,
// W:[512,512] f32.  64x64 tile per block, 8 waves, bf16 WMMA, K-step 32.
// Epilogue scatters bf16 into [B,H,L,DK] layout.  grid.z selects Q/K/V.
// ---------------------------------------------------------------------------
__global__ __launch_bounds__(256) void la_qkv_gemm_kernel(
    const float* __restrict__ X,
    const float* __restrict__ Wq, const float* __restrict__ bq,
    const float* __restrict__ Wk, const float* __restrict__ bk,
    const float* __restrict__ Wv, const float* __restrict__ bv,
    unsigned short* __restrict__ Qo, unsigned short* __restrict__ Ko,
    unsigned short* __restrict__ Vo)
{
    __shared__ __bf16 As[64][40];
    __shared__ __bf16 Bs[64][40];

    const int m0 = blockIdx.x << 6;
    const int n0 = blockIdx.y << 6;
    const int z  = blockIdx.z;
    const float* Wm = (z == 0) ? Wq : (z == 1) ? Wk : Wv;
    const float* bm = (z == 0) ? bq : (z == 1) ? bk : bv;
    unsigned short* dst = (z == 0) ? Qo : (z == 1) ? Ko : Vo;

    const int t    = threadIdx.x;
    const int w    = t >> 5;
    const int wr   = (w & 3) << 4;   // 0,16,32,48
    const int wc   = (w >> 2) << 5;  // 0,32
    const int lane = t & 31;
    const int nl   = lane & 15;
    const int lh   = lane >> 4;

    const int cr = t >> 2;           // copy row 0..63
    const int cc = (t & 3) << 3;     // copy col {0,8,16,24}

    v8f acc0 = {}; v8f acc1 = {};

    for (int kt = 0; kt < 16; ++kt) {
        const int k0 = kt << 5;
        __syncthreads();
        {
            const float* xs = X  + (size_t)(m0 + cr) * DMODEL + k0 + cc;
            const float* ws = Wm + (size_t)(n0 + cr) * DMODEL + k0 + cc;
            __builtin_prefetch(xs + 32, 0, 1);  // next K tile -> global_prefetch_b8
#pragma unroll
            for (int i = 0; i < 8; ++i) {
                As[cr][cc + i] = (__bf16)xs[i];
                Bs[cr][cc + i] = (__bf16)ws[i];
            }
        }
        __syncthreads();
        v16bf a  = ld_frag_row(&As[wr][0], 40);
        v16bf b0 = ld_frag_row(&Bs[wc][0], 40);
        v16bf b1 = ld_frag_row(&Bs[wc + 16][0], 40);
        acc0 = wmma_bf16(a, b0, acc0);
        acc1 = wmma_bf16(a, b1, acc1);
    }

#pragma unroll
    for (int j = 0; j < 2; ++j) {
        v8f c = j ? acc1 : acc0;
        const int col  = n0 + wc + (j << 4) + nl;
        const int h    = col >> 6;
        const int dk   = col & (HDIM - 1);
        const float bi = bm[col];
#pragma unroll
        for (int v = 0; v < 8; ++v) {
            const int row  = m0 + wr + v + (lh << 3);
            const int bidx = row >> 11;          // / SEQ
            const int l    = row & (SEQ - 1);
            const float val = c[v] + bi;
            dst[(((size_t)bidx * NHEAD + h) * SEQ + l) * HDIM + dk] =
                bf_bits((__bf16)val);
        }
    }
}

// ---------------------------------------------------------------------------
// Kernel 2: banded local attention.  One block = (b, h, 64-query tile),
// 4 waves; keys j in [q0-32, q0+95].  K/V tiles staged by the Tensor Data
// Mover (tensor_load_to_lds + s_wait_tensorcnt); WMMA for QK^T and P.V;
// in-register softmax; atomic accumulation of attn/H into the mean plane.
// ---------------------------------------------------------------------------
__global__ __launch_bounds__(128) void la_attn_kernel(
    const unsigned short* __restrict__ Qg,
    const unsigned short* __restrict__ Kg,
    const unsigned short* __restrict__ Vg,
    unsigned short* __restrict__ AO,     // bf16 [B, L, D] (heads concatenated)
    float* __restrict__ meanOut)         // f32 [B, L, L], pre-zeroed
{
    __shared__ __align__(128) __bf16 Qs[64][72];
    __shared__ __align__(128) __bf16 Ks[128][72];
    __shared__ __align__(128) __bf16 Vs[128][72];
    __shared__ __align__(128) __bf16 Ps[64][136];

    const int q0     = blockIdx.x << 6;
    const int h      = blockIdx.y;
    const int bb     = blockIdx.z;
    const int kstart = q0 - 32;

    const size_t headBase = ((size_t)bb * NHEAD + h) * SEQ * HDIM;  // elements
    const int t = threadIdx.x;

    // stage Q tile: 64 x 64 (regular vector loads)
    {
        const int r  = t >> 1;
        const int c0 = (t & 1) << 5;
        const unsigned short* src = Qg + headBase + (size_t)(q0 + r) * HDIM + c0;
#pragma unroll
        for (int i = 0; i < 32; ++i) Qs[r][c0 + i] = bits_bf(src[i]);
    }

    // zero only out-of-band LDS rows (sequence edges); TDM fills the rest.
    {
        const int r = t;                 // 0..127
        const int j = kstart + r;
        if (j < 0 || j >= SEQ) {
#pragma unroll
            for (int i = 0; i < HDIM; ++i) {
                Ks[r][i] = (__bf16)0.0f;
                Vs[r][i] = (__bf16)0.0f;
            }
        }
    }

    // TDM: load the valid key-row window [vstart, vend) for K and V.
    {
        const int vstart = (kstart < 0) ? 0 : kstart;
        const int vend   = (kstart + 128 > SEQ) ? SEQ : (kstart + 128);
        const int nval   = vend - vstart;
        const int r0     = vstart - kstart;       // first LDS row written by TDM

        if (t == 0) {   // one wave issues; TDM ignores EXEC, per-wave TENSORcnt
            const unsigned long long koff =
                (unsigned long long)(headBase + (size_t)vstart * HDIM) * 2ull;
            const unsigned long long kga =
                (unsigned long long)(const char*)Kg + koff;
            const unsigned long long vga =
                (unsigned long long)(const char*)Vg + koff;
            const unsigned klds = (unsigned)(unsigned long long)(size_t)&Ks[r0][0];
            const unsigned vlds = (unsigned)(unsigned long long)(size_t)&Vs[r0][0];
            tdm_load_2d(kga, klds, nval);
            tdm_load_2d(vga, vlds, nval);
            __builtin_amdgcn_s_wait_tensorcnt(0);
        }
    }
    __syncthreads();

    const int w    = t >> 5;      // wave 0..3
    const int qr   = w << 4;      // 16-query sub-tile
    const int lane = t & 31;
    const int nl   = lane & 15;
    const int lh   = lane >> 4;

    // Q fragments for both 32-wide dk chunks (reused across all key tiles)
    v16bf qa0 = ld_frag_row(&Qs[qr][0],  72);
    v16bf qa1 = ld_frag_row(&Qs[qr][32], 72);

    // S = Q K^T over the 128-key band: 8 tiles of 16x16, 2 WMMAs each
    v8f S[8];
#pragma unroll
    for (int kt = 0; kt < 8; ++kt) {
        v16bf kb0 = ld_frag_row(&Ks[kt << 4][0],  72);
        v16bf kb1 = ld_frag_row(&Ks[kt << 4][32], 72);
        v8f c = {};
        c = wmma_bf16(qa0, kb0, c);
        c = wmma_bf16(qa1, kb1, c);
        S[kt] = c;
    }

    // mask + scale (1/sqrt(64)) + row max
    float mx[8];
#pragma unroll
    for (int v = 0; v < 8; ++v) mx[v] = -3.0e38f;
#pragma unroll
    for (int kt = 0; kt < 8; ++kt) {
        const int j = kstart + (kt << 4) + nl;
#pragma unroll
        for (int v = 0; v < 8; ++v) {
            const int q   = q0 + qr + v + (lh << 3);
            const int dlt = q - j;
            const bool ok = (j >= 0) && (j < SEQ) && (dlt <= 32) && (dlt >= -32);
            const float s = ok ? S[kt][v] * 0.125f : -1.0e9f;
            S[kt][v] = s;
            mx[v] = fmaxf(mx[v], s);
        }
    }
    // reduce over the 16 lanes of each half (C-layout rows live per half)
#pragma unroll
    for (int v = 0; v < 8; ++v) {
        mx[v] = fmaxf(mx[v], __shfl_xor(mx[v], 1, 32));
        mx[v] = fmaxf(mx[v], __shfl_xor(mx[v], 2, 32));
        mx[v] = fmaxf(mx[v], __shfl_xor(mx[v], 4, 32));
        mx[v] = fmaxf(mx[v], __shfl_xor(mx[v], 8, 32));
    }
    float sum[8];
#pragma unroll
    for (int v = 0; v < 8; ++v) sum[v] = 0.f;
#pragma unroll
    for (int kt = 0; kt < 8; ++kt) {
#pragma unroll
        for (int v = 0; v < 8; ++v) {
            const float p = __expf(S[kt][v] - mx[v]);
            S[kt][v] = p;
            sum[v] += p;
        }
    }
#pragma unroll
    for (int v = 0; v < 8; ++v) {
        sum[v] += __shfl_xor(sum[v], 1, 32);
        sum[v] += __shfl_xor(sum[v], 2, 32);
        sum[v] += __shfl_xor(sum[v], 4, 32);
        sum[v] += __shfl_xor(sum[v], 8, 32);
    }
    float rinv[8];
#pragma unroll
    for (int v = 0; v < 8; ++v) rinv[v] = 1.0f / sum[v];

    // normalized P -> LDS (re-fragmentation) + mean accumulation (attn / H)
#pragma unroll
    for (int kt = 0; kt < 8; ++kt) {
        const int j = kstart + (kt << 4) + nl;
#pragma unroll
        for (int v = 0; v < 8; ++v) {
            const float p = S[kt][v] * rinv[v];
            const int m = qr + v + (lh << 3);
            Ps[m][(kt << 4) + nl] = (__bf16)p;
            if (j >= 0 && j < SEQ) {
                const int q = q0 + m;
                atomicAdd(meanOut + ((size_t)bb * SEQ + q) * SEQ + j, p * 0.125f);
            }
        }
    }

    // O = P V : contraction over 128 keys, output 16 x 64
    v16bf pa[4];
#pragma unroll
    for (int kc = 0; kc < 4; ++kc) pa[kc] = ld_frag_row(&Ps[qr][kc << 5], 136);

#pragma unroll
    for (int dt = 0; dt < 4; ++dt) {
        v8f o = {};
#pragma unroll
        for (int kc = 0; kc < 4; ++kc) {
            v16bf vb = ld_frag_col(&Vs[kc << 5][dt << 4], 72);
            o = wmma_bf16(pa[kc], vb, o);
        }
#pragma unroll
        for (int v = 0; v < 8; ++v) {
            const int m = qr + v + (lh << 3);
            const int q = q0 + m;
            AO[((size_t)bb * SEQ + q) * DMODEL + (h << 6) + (dt << 4) + nl] =
                bf_bits((__bf16)o[v]);
        }
    }
}

// ---------------------------------------------------------------------------
// Kernel 3: output projection.  out = AO(bf16) @ Wo^T + bo, f32 result.
// ---------------------------------------------------------------------------
__global__ __launch_bounds__(256) void la_out_gemm_kernel(
    const unsigned short* __restrict__ Xbf,
    const float* __restrict__ Wo, const float* __restrict__ bo,
    float* __restrict__ out)
{
    __shared__ __bf16 As[64][40];
    __shared__ __bf16 Bs[64][40];

    const int m0 = blockIdx.x << 6;
    const int n0 = blockIdx.y << 6;

    const int t    = threadIdx.x;
    const int w    = t >> 5;
    const int wr   = (w & 3) << 4;
    const int wc   = (w >> 2) << 5;
    const int lane = t & 31;
    const int nl   = lane & 15;
    const int lh   = lane >> 4;

    const int cr = t >> 2;
    const int cc = (t & 3) << 3;

    v8f acc0 = {}; v8f acc1 = {};

    for (int kt = 0; kt < 16; ++kt) {
        const int k0 = kt << 5;
        __syncthreads();
        {
            const unsigned short* xs = Xbf + (size_t)(m0 + cr) * DMODEL + k0 + cc;
            const float*          ws = Wo  + (size_t)(n0 + cr) * DMODEL + k0 + cc;
#pragma unroll
            for (int i = 0; i < 8; ++i) {
                As[cr][cc + i] = bits_bf(xs[i]);
                Bs[cr][cc + i] = (__bf16)ws[i];
            }
        }
        __syncthreads();
        v16bf a  = ld_frag_row(&As[wr][0], 40);
        v16bf b0 = ld_frag_row(&Bs[wc][0], 40);
        v16bf b1 = ld_frag_row(&Bs[wc + 16][0], 40);
        acc0 = wmma_bf16(a, b0, acc0);
        acc1 = wmma_bf16(a, b1, acc1);
    }

#pragma unroll
    for (int j = 0; j < 2; ++j) {
        v8f c = j ? acc1 : acc0;
        const int col  = n0 + wc + (j << 4) + nl;
        const float bi = bo[col];
#pragma unroll
        for (int v = 0; v < 8; ++v) {
            const int row = m0 + wr + v + (lh << 3);
            out[(size_t)row * DMODEL + col] = c[v] + bi;
        }
    }
}

// ---------------------------------------------------------------------------
// Host launcher
// ---------------------------------------------------------------------------
extern "C" void kernel_launch(void* const* d_in, const int* in_sizes, int n_in,
                              void* d_out, int out_size, void* d_ws, size_t ws_size,
                              hipStream_t stream) {
    (void)in_sizes; (void)n_in; (void)out_size; (void)ws_size;

    const float* x  = (const float*)d_in[0];
    const float* Wq = (const float*)d_in[1];
    const float* bq = (const float*)d_in[2];
    const float* Wk = (const float*)d_in[3];
    const float* bk = (const float*)d_in[4];
    const float* Wv = (const float*)d_in[5];
    const float* bv = (const float*)d_in[6];
    const float* Wo = (const float*)d_in[7];
    const float* bo = (const float*)d_in[8];

    float* out  = (float*)d_out;
    float* mean = out + (size_t)BATCH * SEQ * DMODEL;   // [B, L, L]

    // workspace: Qb | Kb | Vb | AO, each bf16 [4.19M elems] = 8 MiB
    const size_t seg = (size_t)BATCH * NHEAD * SEQ * HDIM * sizeof(unsigned short);
    char* ws = (char*)d_ws;
    unsigned short* Qb = (unsigned short*)(ws);
    unsigned short* Kb = (unsigned short*)(ws + seg);
    unsigned short* Vb = (unsigned short*)(ws + 2 * seg);
    unsigned short* AO = (unsigned short*)(ws + 3 * seg);

    // 1) zero the mean plane (B*L*L floats = 4.19M float4)
    la_zero_kernel<<<dim3(2048), dim3(256), 0, stream>>>(
        (float4*)mean, (int)((size_t)BATCH * SEQ * SEQ / 4));

    // 2) Q,K,V projections: grid (M/64, N/64, 3)
    la_qkv_gemm_kernel<<<dim3((BATCH * SEQ) / 64, DMODEL / 64, 3), dim3(256), 0, stream>>>(
        x, Wq, bq, Wk, bk, Wv, bv, Qb, Kb, Vb);

    // 3) banded attention: grid (L/64, H, B)
    la_attn_kernel<<<dim3(SEQ / 64, NHEAD, BATCH), dim3(128), 0, stream>>>(
        Qb, Kb, Vb, AO, mean);

    // 4) output projection
    la_out_gemm_kernel<<<dim3((BATCH * SEQ) / 64, DMODEL / 64, 1), dim3(256), 0, stream>>>(
        AO, Wo, bo, out);
}
